// WavenetModel_47450798686597
// MI455X (gfx1250) — matrix operand order repaired and checked by reading, not compiled, not run on hardware
//
#include <hip/hip_runtime.h>
#include <hip/hip_bf16.h>

// ---------------------------------------------------------------------------
// WaveNet forward for MI455X (gfx1250), wave32 + v_wmma_f32_16x16x32_bf16,
// with async global->LDS double-buffered staging in the fused head.
// B=4, T=32768, R=D=64, S=C=256, L=30 layers.
// ---------------------------------------------------------------------------

typedef __bf16 bf16_t;
typedef __attribute__((ext_vector_type(8)))  __bf16 bf16x8;
typedef __attribute__((ext_vector_type(16))) __bf16 bf16x16;
typedef __attribute__((ext_vector_type(8)))  float  floatx8;
typedef __attribute__((ext_vector_type(4)))  int    intx4;

#define NB 4
#define NLAY 30
#define TT 64           // time-tile per block

#if defined(__HIP_DEVICE_COMPILE__) && \
    __has_builtin(__builtin_amdgcn_global_load_async_to_lds_b128)
#define HAVE_ASYNC 1
#else
#define HAVE_ASYNC 0
#endif

__device__ __forceinline__ void async_wait_all() {
#if HAVE_ASYNC
#if __has_builtin(__builtin_amdgcn_s_wait_asynccnt)
  __builtin_amdgcn_s_wait_asynccnt(0);
#else
  asm volatile("s_wait_asynccnt 0" ::: "memory");
#endif
#endif
}

__device__ __forceinline__ floatx8 wmma_bf16(bf16x16 a, bf16x16 b, floatx8 c) {
  return __builtin_amdgcn_wmma_f32_16x16x32_bf16(false, a, false, b, (short)0, c,
                                                 false, false);
}

// A fragment (16x32 bf16, ISA 7.12.2): lane half 0: K = k0+{0..7,16..23};
// half 1: K = k0+{8..15,24..31}; row m = caller-fixed per lane (m%16 = lane%16).
__device__ __forceinline__ bf16x16 load_a_frag(const bf16_t* __restrict__ W,
                                               int ld, int m, int k0, int lane) {
  const bf16_t* row = W + (size_t)m * ld + k0 + ((lane >> 4) & 1) * 8;
  bf16x8 lo = *(const bf16x8*)(row);
  bf16x8 hi = *(const bf16x8*)(row + 16);
  return __builtin_shufflevector(lo, hi, 0, 1, 2, 3, 4, 5, 6, 7,
                                 8, 9, 10, 11, 12, 13, 14, 15);
}

// B fragment (32x16 bf16): lane half 0: K = k0+{0..7,16..23}, half 1: K = k0+{8..15,24..31},
// column n fixed per lane. X stored N-major: X[n][k], leading dim ldk (elems).
__device__ __forceinline__ bf16x16 load_b_frag(const bf16_t* X, int ldk, int n,
                                               int k0, int lane) {
  const bf16_t* p = X + (size_t)n * ldk + k0 + ((lane >> 4) & 1) * 8;
  bf16x8 lo = *(const bf16x8*)(p);
  bf16x8 hi = *(const bf16x8*)(p + 16);
  return __builtin_shufflevector(lo, hi, 0, 1, 2, 3, 4, 5, 6, 7,
                                 8, 9, 10, 11, 12, 13, 14, 15);
}

// ---------------------------------------------------------------------------
// Weight prep: fp32 -> bf16, repack dilated-conv taps into one [64 x 128] A.
// ---------------------------------------------------------------------------
__global__ void prep_kernel(const float* __restrict__ filt_w,
                            const float* __restrict__ gate_w,
                            const float* __restrict__ res_w,
                            const float* __restrict__ skip_w,
                            const float* __restrict__ end1_w,
                            const float* __restrict__ end2_w,
                            bf16_t* wf, bf16_t* wg, bf16_t* resW,
                            bf16_t* skipW, bf16_t* e1W, bf16_t* e2W) {
  int idx = blockIdx.x * 256 + threadIdx.x;
  const int NFG = NLAY * 64 * 128;   // 245760
  const int NRES = NLAY * 64 * 64;   // 122880
  const int NSKIP = 256 * NLAY * 64; // 491520
  const int NE = 256 * 256;          // 65536
  if (idx < NFG) {
    int k = idx & 127;                  // 0..63 -> tap0 (t-d), 64..127 -> tap1 (t)
    int dch = (idx >> 7) & 63;
    int i = idx >> 13;
    int tap = k >> 6, r = k & 63;
    int src = ((i * 64 + dch) * 64 + r) * 2 + tap;
    wf[idx] = (bf16_t)filt_w[src];
    wg[idx] = (bf16_t)gate_w[src];
    return;
  }
  idx -= NFG;
  if (idx < NRES) { resW[idx] = (bf16_t)res_w[idx]; return; }
  idx -= NRES;
  if (idx < NSKIP) {                   // skipW[s][i*64+d] = skip_w[i][s][d]
    int col = idx % (NLAY * 64);
    int s = idx / (NLAY * 64);
    int i = col >> 6, dd = col & 63;
    skipW[idx] = (bf16_t)skip_w[(i * 256 + s) * 64 + dd];
    return;
  }
  idx -= NSKIP;
  if (idx < NE) { e1W[idx] = (bf16_t)end1_w[idx]; return; }
  idx -= NE;
  if (idx < NE) { e2W[idx] = (bf16_t)end2_w[idx]; return; }
}

// ---------------------------------------------------------------------------
// Causal 1x1 conv: x[b][r][t] = causal_w[r]*y[b][t] + causal_b[r]
// ---------------------------------------------------------------------------
__global__ __launch_bounds__(256) void causal_kernel(
    const float* __restrict__ y, const float* __restrict__ cw,
    const float* __restrict__ cb, float* __restrict__ xA, int T) {
  int b = blockIdx.y;
  int t = blockIdx.x * 256 + threadIdx.x;
  float v = y[(size_t)b * T + t];
  float* xb = xA + (size_t)b * 64 * T;
#pragma unroll
  for (int r = 0; r < 64; r++) xb[(size_t)r * T + t] = cw[r] * v + cb[r];
}

// ---------------------------------------------------------------------------
// One WaveNet layer over a [64 x TT] time tile.
//   pre_{f,g} = W[64x128] @ [x(t-d); x(t)]     (4 K-blocks of 32)
//   out       = tanh(pre_f) * sigmoid(pre_g)   -> bf16 to out_ws[t][dch] + LDS
//   x_out     = x_in + resW[64x64] @ out       (2 K-blocks)
// ---------------------------------------------------------------------------
__global__ __launch_bounds__(256) void layer_kernel(
    const float* __restrict__ xin, float* __restrict__ xout,
    const bf16_t* __restrict__ wf, const bf16_t* __restrict__ wg,
    const bf16_t* __restrict__ resW, bf16_t* __restrict__ outws,
    int dil, int T) {
  __shared__ bf16_t bxT[TT][136];         // [n=t][k]: k<64 shifted, k>=64 current
  __shared__ float  xcf[64][68];          // current x (fp32) for residual add
  __shared__ float  pre[2][64][68];       // filt / gate pre-activations
  __shared__ bf16_t obfT[TT][72];         // gated output, [n=t][k=dch]

  const int b = blockIdx.y;
  const int t0 = blockIdx.x * TT;
  const int tid = threadIdx.x;
  const int lane = tid & 31;
  const int wave = tid >> 5;
  const int half = (lane >> 4) & 1;
  const int ln = lane & 15;

  const float* xb = xin + (size_t)b * 64 * T;
  for (int idx = tid; idx < 64 * TT; idx += 256) {
    int r = idx / TT, tl = idx % TT;
    int tc = t0 + tl;
    float vc = xb[(size_t)r * T + tc];
    int ts = tc - dil;
    float vs = (ts >= 0) ? xb[(size_t)r * T + ts] : 0.0f;
    bxT[tl][r] = (bf16_t)vs;
    bxT[tl][64 + r] = (bf16_t)vc;
    xcf[r][tl] = vc;
  }
  __syncthreads();

  // ---- filt/gate GEMM: waves 0-3 -> filt rows, waves 4-7 -> gate rows ----
  {
    const int sel = wave >> 2;
    const int mi = wave & 3;
    const bf16_t* W = sel ? wg : wf;
    floatx8 acc[4] = {};
    const int m = mi * 16 + ln;
#pragma unroll
    for (int kb = 0; kb < 4; kb++) {
      bf16x16 a = load_a_frag(W, 128, m, kb * 32, lane);
#pragma unroll
      for (int ni = 0; ni < 4; ni++) {
        bf16x16 bf = load_b_frag(&bxT[0][0], 136, ni * 16 + ln, kb * 32, lane);
        acc[ni] = wmma_bf16(a, bf, acc[ni]);
      }
    }
    const int mrow = mi * 16 + half * 8;
#pragma unroll
    for (int ni = 0; ni < 4; ni++)
#pragma unroll
      for (int k = 0; k < 8; k++) pre[sel][mrow + k][ni * 16 + ln] = acc[ni][k];
  }
  __syncthreads();

  // ---- gated activation, emit bf16 (channel-contiguous global layout) ----
  bf16_t* ob = outws + ((size_t)b * T + t0) * 64;
  for (int idx = tid; idx < 64 * TT; idx += 256) {
    int dch = idx & 63, tl = idx >> 6;   // consecutive threads -> contiguous dch
    float f = tanhf(pre[0][dch][tl]);
    float gz = pre[1][dch][tl];
    float g = 1.0f / (1.0f + expf(-gz));
    bf16_t o = (bf16_t)(f * g);
    obfT[tl][dch] = o;
    ob[(size_t)tl * 64 + dch] = o;
  }
  __syncthreads();

  // ---- residual GEMM + add: 16 tiles over 8 waves (2 each) ----
  {
    floatx8 racc[2] = {};
    const int mi = wave & 3;
    const int nbase = (wave >> 2) * 2;
    const int m = mi * 16 + ln;
#pragma unroll
    for (int kb = 0; kb < 2; kb++) {
      bf16x16 a = load_a_frag(resW, 64, m, kb * 32, lane);
#pragma unroll
      for (int j = 0; j < 2; j++) {
        bf16x16 bf =
            load_b_frag(&obfT[0][0], 72, (nbase + j) * 16 + ln, kb * 32, lane);
        racc[j] = wmma_bf16(a, bf, racc[j]);
      }
    }
    float* xo = xout + (size_t)b * 64 * T + t0;
    const int mr = mi * 16 + half * 8;
#pragma unroll
    for (int j = 0; j < 2; j++) {
      int nc = (nbase + j) * 16 + ln;
#pragma unroll
      for (int k = 0; k < 8; k++)
        xo[(size_t)(mr + k) * T + nc] = racc[j][k] + xcf[mr + k][nc];
    }
  }
}

// ---------------------------------------------------------------------------
// Fused head: skip = skipW[256x1920] @ outs[1920xTT]  (K over 30 layers x 64)
//             h = relu(skip); h = relu(e1W@h + b1); logits = e2W@h + b2
//             out[b][t][c] = logits[c][t]
// K-slabs of 32 are double-buffered in LDS via GLOBAL_LOAD_ASYNC_TO_LDS_B128
// (ASYNCcnt), so slab kb+1 streams while slab kb feeds the WMMAs.
// ---------------------------------------------------------------------------
__global__ __launch_bounds__(256) void final_kernel(
    const bf16_t* __restrict__ outws,  // [30][B][T][64]
    const bf16_t* __restrict__ skipW,  // [256][1920]
    const bf16_t* __restrict__ e1W, const float* __restrict__ e1b,
    const bf16_t* __restrict__ e2W, const float* __restrict__ e2b,
    float* __restrict__ out, int T) {
  __shared__ bf16_t xsT[2][TT][40];  // double-buffered K-slab, [n=t][k], 80B rows
  __shared__ bf16_t hT[TT][264];     // activations [n=t][k=chan]

  const int b = blockIdx.y;
  const int t0 = blockIdx.x * TT;
  const int tid = threadIdx.x;
  const int lane = tid & 31;
  const int wave = tid >> 5;
  const int half = (lane >> 4) & 1;
  const int ln = lane & 15;

  // staging decomposition: 64 t-rows x 64B; one 16B chunk per thread
  const int tl_s = tid >> 2;
  const int c_s = tid & 3;

  auto issue_slab = [&](int kb, int buf) {
    const int K0 = kb * 32;
    const int li = K0 >> 6, dch0 = K0 & 63;
    const bf16_t* g = outws +
        (((size_t)li * NB + b) * T + (t0 + tl_s)) * 64 + dch0 + c_s * 8;
    bf16_t* l = &xsT[buf][tl_s][c_s * 8];
#if HAVE_ASYNC
    __builtin_amdgcn_global_load_async_to_lds_b128(
        (__attribute__((address_space(1))) intx4*)g,
        (__attribute__((address_space(3))) intx4*)l, 0, 0);
#else
    *(bf16x8*)l = *(const bf16x8*)g;
#endif
  };

  // each wave owns output rows [32*wave, 32*wave+32) as 2 row-tiles x 4 col-tiles
  floatx8 acc[8] = {};
  issue_slab(0, 0);
  for (int kb = 0; kb < 60; kb++) {
    const int K0 = kb * 32;
    async_wait_all();       // this wave's slab-kb copies have landed in LDS
    __syncthreads();        // everyone's landed; prev compute done -> buf reusable
    if (kb + 1 < 60) issue_slab(kb + 1, (kb + 1) & 1);
    const bf16_t* xbuf = &xsT[kb & 1][0][0];
    bf16x16 a0 = load_a_frag(skipW, 1920, (2 * wave + 0) * 16 + ln, K0, lane);
    bf16x16 a1 = load_a_frag(skipW, 1920, (2 * wave + 1) * 16 + ln, K0, lane);
#pragma unroll
    for (int ni = 0; ni < 4; ni++) {
      bf16x16 bf = load_b_frag(xbuf, 40, ni * 16 + ln, 0, lane);
      acc[ni] = wmma_bf16(a0, bf, acc[ni]);
      acc[4 + ni] = wmma_bf16(a1, bf, acc[4 + ni]);
    }
  }
  __syncthreads();
  // relu(skip) -> bf16 LDS
#pragma unroll
  for (int r = 0; r < 2; r++) {
    int mr = (2 * wave + r) * 16 + half * 8;
#pragma unroll
    for (int ni = 0; ni < 4; ni++) {
      int nc = ni * 16 + ln;
#pragma unroll
      for (int k = 0; k < 8; k++)
        hT[nc][mr + k] = (bf16_t)fmaxf(acc[4 * r + ni][k], 0.0f);
    }
  }
  __syncthreads();

  // ---- end1: 256x256, K in 8 blocks ----
  floatx8 acc2[8] = {};
#pragma unroll
  for (int kb = 0; kb < 8; kb++) {
    bf16x16 a0 = load_a_frag(e1W, 256, (2 * wave + 0) * 16 + ln, kb * 32, lane);
    bf16x16 a1 = load_a_frag(e1W, 256, (2 * wave + 1) * 16 + ln, kb * 32, lane);
#pragma unroll
    for (int ni = 0; ni < 4; ni++) {
      bf16x16 bf = load_b_frag(&hT[0][0], 264, ni * 16 + ln, kb * 32, lane);
      acc2[ni] = wmma_bf16(a0, bf, acc2[ni]);
      acc2[4 + ni] = wmma_bf16(a1, bf, acc2[4 + ni]);
    }
  }
  __syncthreads();  // all reads of hT done before overwrite
#pragma unroll
  for (int r = 0; r < 2; r++) {
    int mr = (2 * wave + r) * 16 + half * 8;
#pragma unroll
    for (int ni = 0; ni < 4; ni++) {
      int nc = ni * 16 + ln;
#pragma unroll
      for (int k = 0; k < 8; k++)
        hT[nc][mr + k] = (bf16_t)fmaxf(acc2[4 * r + ni][k] + e1b[mr + k], 0.0f);
    }
  }
  __syncthreads();

  // ---- end2: 256x256 -> logits, write [b][t][c] ----
  floatx8 acc3[8] = {};
#pragma unroll
  for (int kb = 0; kb < 8; kb++) {
    bf16x16 a0 = load_a_frag(e2W, 256, (2 * wave + 0) * 16 + ln, kb * 32, lane);
    bf16x16 a1 = load_a_frag(e2W, 256, (2 * wave + 1) * 16 + ln, kb * 32, lane);
#pragma unroll
    for (int ni = 0; ni < 4; ni++) {
      bf16x16 bf = load_b_frag(&hT[0][0], 264, ni * 16 + ln, kb * 32, lane);
      acc3[ni] = wmma_bf16(a0, bf, acc3[ni]);
      acc3[4 + ni] = wmma_bf16(a1, bf, acc3[4 + ni]);
    }
  }
#pragma unroll
  for (int r = 0; r < 2; r++) {
    int mr = (2 * wave + r) * 16 + half * 8;
#pragma unroll
    for (int ni = 0; ni < 4; ni++) {
      int nc = ni * 16 + ln;
      float* op = out + ((size_t)b * T + t0 + nc) * 256 + mr;
#pragma unroll
      for (int k = 0; k < 8; k++) op[k] = acc3[4 * r + ni][k] + e2b[mr + k];
    }
  }
}

// ---------------------------------------------------------------------------
extern "C" void kernel_launch(void* const* d_in, const int* in_sizes, int n_in,
                              void* d_out, int out_size, void* d_ws,
                              size_t ws_size, hipStream_t stream) {
  const float* y = (const float*)d_in[0];
  const float* causal_w = (const float*)d_in[1];
  const float* causal_b = (const float*)d_in[2];
  const float* filt_w = (const float*)d_in[3];
  const float* gate_w = (const float*)d_in[4];
  const float* res_w = (const float*)d_in[5];
  const float* skip_w = (const float*)d_in[6];
  const float* end1_w = (const float*)d_in[7];
  const float* end1_b = (const float*)d_in[8];
  const float* end2_w = (const float*)d_in[9];
  const float* end2_b = (const float*)d_in[10];
  float* out = (float*)d_out;

  const int T = in_sizes[0] / NB;  // 32768

  // ---- workspace layout ----
  size_t off = 0;
  auto alloc = [&](size_t bytes) {
    size_t o = off;
    off = (off + bytes + 255) & ~(size_t)255;
    return o;
  };
  char* ws = (char*)d_ws;
  size_t xBytes = (size_t)NB * 64 * T * sizeof(float);            // 32 MB
  float* xA = (float*)(ws + alloc(xBytes));
  float* xB = (float*)(ws + alloc(xBytes));
  bf16_t* outws = (bf16_t*)(ws + alloc((size_t)NLAY * NB * 64 * T * 2));  // 480 MB
  bf16_t* wf = (bf16_t*)(ws + alloc((size_t)NLAY * 64 * 128 * 2));
  bf16_t* wg = (bf16_t*)(ws + alloc((size_t)NLAY * 64 * 128 * 2));
  bf16_t* resW = (bf16_t*)(ws + alloc((size_t)NLAY * 64 * 64 * 2));
  bf16_t* skipW = (bf16_t*)(ws + alloc((size_t)256 * NLAY * 64 * 2));
  bf16_t* e1W = (bf16_t*)(ws + alloc((size_t)256 * 256 * 2));
  bf16_t* e2W = (bf16_t*)(ws + alloc((size_t)256 * 256 * 2));
  (void)ws_size;

  // 1) weight conversion / packing
  {
    int total = NLAY * 64 * 128 + NLAY * 64 * 64 + 256 * NLAY * 64 + 2 * 256 * 256;
    prep_kernel<<<(total + 255) / 256, 256, 0, stream>>>(
        filt_w, gate_w, res_w, skip_w, end1_w, end2_w, wf, wg, resW, skipW,
        e1W, e2W);
  }
  // 2) causal expansion 1 -> 64
  causal_kernel<<<dim3(T / 256, NB), 256, 0, stream>>>(y, causal_w, causal_b,
                                                       xA, T);
  // 3) 30 residual layers (ping-pong x)
  for (int i = 0; i < NLAY; i++) {
    int d = 1 << (i % 10);
    const float* xin = (i & 1) ? xB : xA;
    float* xout = (i & 1) ? xA : xB;
    layer_kernel<<<dim3(T / TT, NB), 256, 0, stream>>>(
        xin, xout, wf + (size_t)i * 64 * 128, wg + (size_t)i * 64 * 128,
        resW + (size_t)i * 64 * 64, outws + (size_t)i * NB * 64 * T, d, T);
  }
  // 4) fused skip reduction + head (async double-buffered K-slabs)
  final_kernel<<<dim3(T / TT, NB), 256, 0, stream>>>(outws, skipW, e1W, end1_b,
                                                     e2W, end2_b, out, T);
}
